// HetNetGNN_42580305772711
// MI455X (gfx1250) — matrix operand order repaired and checked by reading, not compiled, not run on hardware
//
#include <hip/hip_runtime.h>
#include <hip/hip_bf16.h>

// ---------------------------------------------------------------------------
// HetNet GNN (cell-free MIMO power allocation), MI455X / gfx1250.
//
// Structure exploited (edge order is s*2048 + u*32 + a, so ei_up/ei_down are
// never read):
//   * per-edge MLPs (1->16->32) dominate FLOPs -> second layer mapped to
//     v_wmma_f32_16x16x32_f16: A = 16 edges x 16 hidden (K padded to 32),
//     two WMMAs (B0/B1) cover the 32 outputs and SHARE one A fragment.
//   * relu(node_msg) terms are summed once per sample (linearity of the
//     segment sum AFTER the per-branch ReLUs) -> no per-edge gathers.
//   * every wave keeps its post-ReLU segment sum in registers across its 32
//     reduction iterations; waves 0-3 = uplink tiles, waves 4-7 = downlink
//     tiles -> perfectly balanced, branch-free, exec-full inner loops
//     (WMMA requires EXEC all ones).
// One 256-thread (8-wave) workgroup per sample; ~42KB LDS of 320KB/WGP.
// ---------------------------------------------------------------------------

typedef __attribute__((ext_vector_type(16))) _Float16 v16h;
typedef __attribute__((ext_vector_type(8)))  float    v8f;

#define NS   1024              // samples
#define NU   64                // UEs per sample
#define NA   32                // APs per sample
#define NUE  (NS*NU)
#define NAP  (NS*NA)
#define EPS  (NU*NA)           // edges per sample = 2048
#define ETOT (NS*EPS)
#define DNPAD 65               // downlink eft row pitch (bank-conflict pad)

struct MlpP { const float* W1; const float* b1; const float* W2; const float* b2; };

struct LayerArgs {
  const float* x_ue;  const float* x_ap;
  const float* ea_up; const float* ea_down;
  MlpP msg_ue, msg_ap, edge_up, edge_down, upd_ue, upd_ap, pow_ue, pow_ap;
  float* x_ue_out; float* x_ap_out;
  int msg_w;        // msg_ue input width: 1 (Ue2Ap) or 2 (PowerConv)
  int has_msg_ap;   // 0 (Ue2Ap) or 1 (PowerConv)
};

__device__ __forceinline__ float sigmoidf_(float z) { return 1.f / (1.f + __expf(-z)); }

// in[w] -> 16 (ReLU) -> 32 (ReLU)
__device__ __forceinline__ void mlp2_node(const float* in, int w, const MlpP& p, float* out32) {
  float h[16];
#pragma unroll
  for (int j = 0; j < 16; ++j) {
    float s = p.b1[j];
    for (int i = 0; i < w; ++i) s += in[i] * p.W1[i * 16 + j];
    h[j] = fmaxf(s, 0.f);
  }
#pragma unroll 4
  for (int f = 0; f < 32; ++f) {
    float s = p.b2[f];
#pragma unroll
    for (int j = 0; j < 16; ++j) s += h[j] * p.W2[j * 32 + f];
    out32[f] = fmaxf(s, 0.f);
  }
}

// 32 -> 16 (ReLU) -> 1 (sigmoid)
__device__ __forceinline__ float pmlp_node(const float* in32, const MlpP& p) {
  float h[16];
#pragma unroll 4
  for (int j = 0; j < 16; ++j) {
    float s = p.b1[j];
#pragma unroll
    for (int i = 0; i < 32; ++i) s += in32[i] * p.W1[i * 16 + j];
    h[j] = fmaxf(s, 0.f);
  }
  float s = p.b2[0];
#pragma unroll
  for (int j = 0; j < 16; ++j) s += h[j] * p.W2[j];
  return sigmoidf_(s);
}

// One workgroup per sample: one full message-passing layer (Ue2Ap/PowerConv).
__global__ __launch_bounds__(256) void layer_kernel(LayerArgs g) {
  // single eft buffer: uplink [u*32+a] at 0, downlink transposed [a][u] at EPS
  __shared__ float eftbuf[EPS + NA * DNPAD];
  __shared__ float msgsum_ue[32];
  __shared__ float msgsum_ap[32];
  __shared__ float updue[NU * 33];
  __shared__ float updap[NA * 33];
  __shared__ float aggap[NA * 33];   // sum over u of relu(edge MLP), 2 waves/AP-tile
  __shared__ float aggue[NU * 33];   // sum over a of relu(edge MLP), 1 wave/UE-tile

  const int s    = blockIdx.x;
  const int tid  = threadIdx.x;
  const int lane = tid & 31;
  const int wave = tid >> 5;
  const int half = lane >> 4;         // WMMA A half: 0 -> hidden 0-7, 1 -> 8-15
  const int l15  = lane & 15;

  if (tid < 32) { msgsum_ue[tid] = 0.f; msgsum_ap[tid] = 0.f; }
  for (int i = tid; i < NA * 33; i += 256) aggap[i] = 0.f;

  // ---- stage edge effects eft = e0*(1-e1) into LDS -------------------------
  const float* eu = g.ea_up   + (size_t)s * EPS * 2;
  const float* ed = g.ea_down + (size_t)s * EPS * 2;
  for (int i = tid; i < EPS; i += 256) {
    __builtin_prefetch(&eu[(i + 256) * 2], 0, 0);          // global_prefetch
    float2 au = ((const float2*)eu)[i];
    float2 ad = ((const float2*)ed)[i];
    eftbuf[i] = au.x * (1.f - au.y);
    eftbuf[EPS + (i & 31) * DNPAD + (i >> 5)] = ad.x * (1.f - ad.y);
  }
  __syncthreads();

  // ---- stage 0: node MLPs (64 UE msgs, 64 UE upds, 32 AP upds, 32 AP msgs) -
  if (tid < NU) {
    int gu = s * NU + tid;
    float in[2] = { g.x_ue[gu * 2 + 0], g.x_ue[gu * 2 + 1] };
    float m[32];
    mlp2_node(in, g.msg_w, g.msg_ue, m);
    for (int f = 0; f < 32; ++f) atomicAdd(&msgsum_ue[f], m[f]);   // ds_add_f32
  } else if (tid < 2 * NU) {
    int u = tid - NU, gu = s * NU + u;
    float in[2] = { g.x_ue[gu * 2 + 0], g.x_ue[gu * 2 + 1] };
    mlp2_node(in, 2, g.upd_ue, &updue[u * 33]);
  } else if (tid < 2 * NU + NA) {
    int a = tid - 2 * NU;
    float in[1] = { g.x_ap[s * NA + a] };
    mlp2_node(in, 1, g.upd_ap, &updap[a * 33]);
  } else if (g.has_msg_ap && tid < 2 * NU + 2 * NA) {
    int a = tid - 2 * NU - NA;
    float in[1] = { g.x_ap[s * NA + a] };
    float m[32];
    mlp2_node(in, 1, g.msg_ap, m);
    for (int f = 0; f < 32; ++f) atomicAdd(&msgsum_ap[f], m[f]);
  }
  __syncthreads();

  // ---- stage 1: per-edge MLP second layer via WMMA + register reduction ----
  // Static schedule, 32 iterations x 2 WMMAs for every wave:
  //   waves 0..3: uplink,  atile = wave&1, u-half = wave>>1
  //   waves 4..7: downlink, utile = wave-4
  {
    const bool up    = (wave < 4);
    const MlpP mp    = up ? g.edge_up : g.edge_down;     // wave-uniform
    const int  atile = wave & 1;
    const int  uhalf = (wave >> 1) & 1;
    const int  utile = wave - 4;

    int base, stride;
    if (up) { base = uhalf * (32 * NA) + atile * 16 + l15; stride = NA; }
    else    { base = EPS + utile * 16 + l15;               stride = DNPAD; }

    // first-layer weights for this lane's 8 hidden units (K = half*8+i)
    float w1r[8], b1r[8];
#pragma unroll
    for (int i = 0; i < 8; ++i) { w1r[i] = mp.W1[half * 8 + i]; b1r[i] = mp.b1[half * 8 + i]; }

    // B0/B1 = W2 columns 0..15 / 16..31 as f16; elems 0..7 = K half*8+k, elems 8..15 stay zero (K pad)
    v16h B0 = {}, B1 = {};
    {
#pragma unroll
      for (int k = 0; k < 8; ++k) {
        B0[k] = (_Float16)mp.W2[(half * 8 + k) * 32 + l15];
        B1[k] = (_Float16)mp.W2[(half * 8 + k) * 32 + 16 + l15];
      }
    }
    const float b20 = mp.b2[l15];
    const float b21 = mp.b2[16 + l15];

    v8f acc0 = {}, acc1 = {};
    for (int it = 0; it < 32; ++it) {
      float e = eftbuf[base + it * stride];                // one ds_load_b32
      v16h Am = {};
#pragma unroll
      for (int i = 0; i < 8; ++i) Am[i] = (_Float16)fmaxf(w1r[i] * e + b1r[i], 0.f);
      v8f z = {};
      v8f D0 = __builtin_amdgcn_wmma_f32_16x16x32_f16(
          false, Am, false, B0, (short)0, z, false, false);
      v8f D1 = __builtin_amdgcn_wmma_f32_16x16x32_f16(
          false, Am, false, B1, (short)0, z, false, false);
#pragma unroll
      for (int i = 0; i < 8; ++i) acc0[i] += fmaxf(D0[i] + b20, 0.f);
#pragma unroll
      for (int i = 0; i < 8; ++i) acc1[i] += fmaxf(D1[i] + b21, 0.f);
    }

    if (up) {          // two waves contribute per AP tile -> ds_add_f32
#pragma unroll
      for (int i = 0; i < 8; ++i) {
        int a = atile * 16 + half * 8 + i;                 // D row -> AP
        atomicAdd(&aggap[a * 33 + l15],      acc0[i]);
        atomicAdd(&aggap[a * 33 + 16 + l15], acc1[i]);
      }
    } else {           // unique writer per (u, feature)
#pragma unroll
      for (int i = 0; i < 8; ++i) {
        int u = utile * 16 + half * 8 + i;                 // D row -> UE
        aggue[u * 33 + l15]      = acc0[i];
        aggue[u * 33 + 16 + l15] = acc1[i];
      }
    }
  }
  __syncthreads();

  // ---- stage 2: node updates (sigmoid power MLPs) --------------------------
  if (tid < NA) {
    int a = tid;
    float v[32];
#pragma unroll 8
    for (int f = 0; f < 32; ++f)
      v[f] = (aggap[a * 33 + f] + msgsum_ue[f]) * (1.f / NU) + updap[a * 33 + f];
    g.x_ap_out[s * NA + a] = pmlp_node(v, g.pow_ap);
  } else if (tid < NA + NU) {
    int u = tid - NA, gu = s * NU + u;
    float v[32];
#pragma unroll 8
    for (int f = 0; f < 32; ++f) {
      float madd = g.has_msg_ap ? msgsum_ap[f] : 0.f;
      v[f] = (aggue[u * 33 + f] + madd) * (1.f / NA) + updue[u * 33 + f];
    }
    float pw = pmlp_node(v, g.pow_ue);
    g.x_ue_out[gu * 2 + 0] = g.x_ue[gu * 2 + 0];   // feature 0 preserved
    g.x_ue_out[gu * 2 + 1] = pw;
  }
}

// EdgeConv: per-edge 5->16->1 logit; sigmoid+softmax are monotonic, so the
// one-hot is argmax of the raw logit over each UE's 32 contiguous edges.
__global__ __launch_bounds__(256) void edge_kernel(
    const float* __restrict__ x_ue, const float* __restrict__ x_ap,
    const float* __restrict__ ea_up,
    const float* __restrict__ W1, const float* __restrict__ b1,
    const float* __restrict__ W2, const float* __restrict__ b2,
    float* __restrict__ ea_up_out) {
  int gid = blockIdx.x * 256 + threadIdx.x;   // one thread per UE (set of 32)
  if (gid >= NUE) return;
  int s = gid / NU;
  float xu0 = x_ue[gid * 2], xu1 = x_ue[gid * 2 + 1];
  int   ebase = gid * NA;
  int   amax = 0;
  float zmax = -3.4e38f;
  float b20 = b2[0];
  for (int a = 0; a < NA; ++a) {
    int e = ebase + a;
    float in[5] = { xu0, xu1, x_ap[s * NA + a], ea_up[e * 2], ea_up[e * 2 + 1] };
    float z = b20;
#pragma unroll
    for (int j = 0; j < 16; ++j) {
      float h = b1[j];
#pragma unroll
      for (int i = 0; i < 5; ++i) h += in[i] * W1[i * 16 + j];
      z += fmaxf(h, 0.f) * W2[j];
    }
    if (z > zmax) { zmax = z; amax = a; }
    ea_up_out[e * 2] = in[3];                 // column 0 preserved
  }
  for (int a = 0; a < NA; ++a)
    ea_up_out[(ebase + a) * 2 + 1] = (a == amax) ? 1.f : 0.f;
}

extern "C" void kernel_launch(void* const* d_in, const int* in_sizes, int n_in,
                              void* d_out, int out_size, void* d_ws, size_t ws_size,
                              hipStream_t stream) {
  (void)in_sizes; (void)n_in; (void)out_size; (void)d_ws; (void)ws_size;
  const float* x_ue    = (const float*)d_in[0];
  const float* x_ap    = (const float*)d_in[1];
  const float* ea_up   = (const float*)d_in[2];
  const float* ea_down = (const float*)d_in[3];
  // params flatten in setup_inputs insertion order, 4 leaves (W1,b1,W2,b2)
  // per MLP: ue2ap @4..31 {msg_ue,edge_up,edge_down,upd_ue,upd_ap,pow_ue,pow_ap},
  // power @32..63 {msg_ue,msg_ap,edge_up,edge_down,upd_ue,upd_ap,pow_ue,pow_ap},
  // edge @64..67. ei_up/ei_down (@68,69) are analytic -> unused.
  auto mk = [&](int b) {
    return MlpP{ (const float*)d_in[b],     (const float*)d_in[b + 1],
                 (const float*)d_in[b + 2], (const float*)d_in[b + 3] };
  };

  float* out   = (float*)d_out;
  float* xue_o = out;                                   // [NUE,2]
  float* xap_o = out + (size_t)NUE * 2;                 // [NAP]
  float* eau_o = xap_o + NAP;                           // [E,2]
  float* ead_o = eau_o + (size_t)ETOT * 2;              // [E,2]

  LayerArgs la{};
  la.x_ue = x_ue; la.x_ap = x_ap; la.ea_up = ea_up; la.ea_down = ea_down;
  la.msg_ue = mk(4);  la.msg_ap = MlpP{nullptr, nullptr, nullptr, nullptr};
  la.edge_up = mk(8); la.edge_down = mk(12);
  la.upd_ue = mk(16); la.upd_ap = mk(20);
  la.pow_ue = mk(24); la.pow_ap = mk(28);
  la.x_ue_out = xue_o; la.x_ap_out = xap_o;
  la.msg_w = 1; la.has_msg_ap = 0;
  hipLaunchKernelGGL(layer_kernel, dim3(NS), dim3(256), 0, stream, la);

  LayerArgs lb{};
  lb.x_ue = xue_o; lb.x_ap = xap_o; lb.ea_up = ea_up; lb.ea_down = ea_down;
  lb.msg_ue = mk(32); lb.msg_ap = mk(36);
  lb.edge_up = mk(40); lb.edge_down = mk(44);
  lb.upd_ue = mk(48); lb.upd_ap = mk(52);
  lb.pow_ue = mk(56); lb.pow_ap = mk(60);
  lb.x_ue_out = xue_o; lb.x_ap_out = xap_o;   // in-place: per-sample WG + barriers
  lb.msg_w = 2; lb.has_msg_ap = 1;
  hipLaunchKernelGGL(layer_kernel, dim3(NS), dim3(256), 0, stream, lb);

  edge_kernel<<<NUE / 256, 256, 0, stream>>>(
      xue_o, xap_o, ea_up,
      (const float*)d_in[64], (const float*)d_in[65],
      (const float*)d_in[66], (const float*)d_in[67], eau_o);

  // ea_down passes through unchanged
  hipMemcpyAsync(ead_o, ea_down, (size_t)ETOT * 2 * sizeof(float),
                 hipMemcpyDeviceToDevice, stream);
}
